// MoEBlock_65712999629236
// MI455X (gfx1250) — matrix operand order repaired and checked by reading, not compiled, not run on hardware
//
#include <hip/hip_runtime.h>
#include <hip/hip_bf16.h>

#define T_TOK   8192
#define H_DIM   1024
#define I_DIM   2816
#define E_NUM   8
#define BM      64
#define BN      128
#define BK      64      // K per LDS stage = 2 WMMA k-steps

typedef __bf16          v16bf  __attribute__((ext_vector_type(16)));
typedef float           v8f    __attribute__((ext_vector_type(8)));
typedef unsigned int    u32x4  __attribute__((ext_vector_type(4)));
typedef float           f32x4  __attribute__((ext_vector_type(4)));
typedef unsigned short  u16x4  __attribute__((ext_vector_type(4)));

union Frag16 { v16bf v; u32x4 q[2]; };

// fp32 -> bf16 round-to-nearest-even
__device__ __forceinline__ unsigned short f2bf(float f) {
  unsigned u = __builtin_bit_cast(unsigned, f);
  unsigned r = u + 0x7FFFu + ((u >> 16) & 1u);
  return (unsigned short)(r >> 16);
}

__device__ __forceinline__ float fast_rcp(float x) {
#if __has_builtin(__builtin_amdgcn_rcpf)
  return __builtin_amdgcn_rcpf(x);
#else
  return 1.0f / x;
#endif
}

// ---- CDNA5 async global->LDS copy (ISA 15.18.3 op 98), ASYNCcnt-tracked ----
__device__ __forceinline__ unsigned lds32(const void* p) {
  return (unsigned)(unsigned long long)p;     // low 32 bits == LDS byte address
}
__device__ __forceinline__ void async_b128(unsigned lds_addr, const void* gptr) {
  asm volatile("global_load_async_to_lds_b128 %0, %1, off"
               :: "v"(lds_addr), "v"((unsigned long long)gptr)
               : "memory");
}
// wait until at most N async ops remain (in-order completion => previous stage done)
#define ASYNC_WAIT(n) asm volatile("s_wait_asynccnt " #n ::: "memory")

// ---------------------------------------------------------------- init
__global__ void k_init(float* __restrict__ out, int n, int* __restrict__ counts) {
  int i = blockIdx.x * blockDim.x + threadIdx.x;
  if (i < E_NUM) counts[i] = 0;
  int stride = gridDim.x * blockDim.x;
  u32x4* o4 = (u32x4*)out;
  int n4 = n >> 2;
  for (int idx = i; idx < n4; idx += stride) o4[idx] = (u32x4)0u;
}

// ---------------------------------------------------------------- fp32 -> bf16 (x)
__global__ void k_cvt(const float* __restrict__ src, unsigned short* __restrict__ dst, int n) {
  int stride = gridDim.x * blockDim.x;
  int n4 = n >> 2;
  const f32x4* s4 = (const f32x4*)src;
  u16x4* d4 = (u16x4*)dst;
  for (int i = blockIdx.x * blockDim.x + threadIdx.x; i < n4; i += stride) {
    f32x4 v = s4[i];
    u16x4 o;
    o[0] = f2bf(v[0]); o[1] = f2bf(v[1]); o[2] = f2bf(v[2]); o[3] = f2bf(v[3]);
    d4[i] = o;
  }
}

// ----------------------------------------------- fp32 [E][R][C] -> bf16 [E][C][R]
__global__ void k_cvt_tr(const float* __restrict__ src, unsigned short* __restrict__ dst,
                         int R, int C) {
  __shared__ unsigned short tile[32][33];
  const int e  = blockIdx.z;
  const int c0 = blockIdx.x * 32;
  const int r0 = blockIdx.y * 32;
  const float* s = src + (size_t)e * R * C;
  unsigned short* d = dst + (size_t)e * C * R;
  for (int i = threadIdx.x; i < 32 * 32; i += 256) {
    int rr = i >> 5, cc = i & 31;
    tile[cc][rr] = f2bf(s[(size_t)(r0 + rr) * C + c0 + cc]);
  }
  __syncthreads();
  for (int i = threadIdx.x; i < 32 * 32; i += 256) {
    int cc = i >> 5, rr = i & 31;
    d[(size_t)(c0 + cc) * R + r0 + rr] = tile[cc][rr];
  }
}

// ---------------------------------------------------------------- router (one wave32 / token)
__global__ void k_router(const float* __restrict__ x, const float* __restrict__ gw,
                         float* __restrict__ logits, float* __restrict__ wew,
                         int* __restrict__ counts, int* __restrict__ list) {
  int tok  = (blockIdx.x * blockDim.x + threadIdx.x) >> 5;
  int lane = threadIdx.x & 31;
  if (tok >= T_TOK) return;
  const float* xr = x + (size_t)tok * H_DIM;

  float acc[E_NUM];
#pragma unroll
  for (int e = 0; e < E_NUM; e++) acc[e] = 0.0f;
  for (int k = lane; k < H_DIM; k += 32) {
    float xv = xr[k];
    const float* g = gw + (size_t)k * E_NUM;
#pragma unroll
    for (int e = 0; e < E_NUM; e++) acc[e] += xv * g[e];
  }
#pragma unroll
  for (int off = 16; off >= 1; off >>= 1) {
#pragma unroll
    for (int e = 0; e < E_NUM; e++) acc[e] += __shfl_xor(acc[e], off, 32);
  }
  if (lane == 0) {
    float mx = acc[0];
#pragma unroll
    for (int e = 1; e < E_NUM; e++) mx = fmaxf(mx, acc[e]);
    float p[E_NUM], s = 0.0f;
#pragma unroll
    for (int e = 0; e < E_NUM; e++) { p[e] = __expf(acc[e] - mx); s += p[e]; }
#pragma unroll
    for (int e = 0; e < E_NUM; e++) p[e] /= s;
    int a = 0;
#pragma unroll
    for (int e = 1; e < E_NUM; e++) if (p[e] > p[a]) a = e;
    int b = (a == 0) ? 1 : 0;
#pragma unroll
    for (int e = 0; e < E_NUM; e++) if (e != a && p[e] > p[b]) b = e;
    float den = p[a] + p[b];
    float wa = p[a] / den, wb = p[b] / den;
#pragma unroll
    for (int e = 0; e < E_NUM; e++) logits[(size_t)tok * E_NUM + e] = acc[e];
#pragma unroll
    for (int e = 0; e < E_NUM; e++)
      wew[(size_t)tok * E_NUM + e] = (e == a) ? wa : ((e == b) ? wb : 0.0f);
    int pa = atomicAdd(&counts[a], 1); list[a * T_TOK + pa] = tok;
    int pb = atomicAdd(&counts[b], 1); list[b * T_TOK + pb] = tok;
  }
}

// ---------------------------------------------------------------- prefix sum (8 values)
__global__ void k_offsets(const int* __restrict__ counts, int* __restrict__ offsets) {
  if (blockIdx.x == 0 && threadIdx.x == 0) {
    int run = 0;
    for (int e = 0; e < E_NUM; e++) { offsets[e] = run; run += counts[e]; }
  }
}

// ---------------------------------------------------------------- GEMM1: h = silu(X Wg) * (X Wu)
// wgT/wuT: [E][I][H] bf16 (N-major, K contiguous). Double-buffered async pipeline.
__global__ void __launch_bounds__(256)
k_gemm1(const unsigned short* __restrict__ xb,
        const unsigned short* __restrict__ wgT,
        const unsigned short* __restrict__ wuT,
        unsigned short* __restrict__ hbuf,
        const int* __restrict__ counts,
        const int* __restrict__ offsets,
        const int* __restrict__ list) {
  const int e  = blockIdx.z;
  const int mt = blockIdx.x;
  const int nt = blockIdx.y;
  const int Ne = counts[e];
  if (mt * BM >= Ne) return;

  __shared__ __align__(16) unsigned short lA [2][BM][BK];   // 2 x 8 KB
  __shared__ __align__(16) unsigned short lBg[2][BN][BK];   // 2 x 16 KB
  __shared__ __align__(16) unsigned short lBu[2][BN][BK];   // 2 x 16 KB
  __shared__ int rows[BM];

  const int tid  = threadIdx.x;
  const int lane = tid & 31;
  const int w    = tid >> 5;
  const int wr   = w >> 2;
  const int wc   = w & 3;
  const int l16  = lane & 15;
  const int lhi  = lane >> 4;

  if (tid < BM) {
    int r = mt * BM + tid;
    rows[tid] = list[e * T_TOK + min(r, Ne - 1)];   // clamp: garbage rows masked later
  }
  __syncthreads();

  v8f accG[2][2], accU[2][2];
#pragma unroll
  for (int i = 0; i < 2; i++)
#pragma unroll
    for (int j = 0; j < 2; j++) { accG[i][j] = (v8f)0.0f; accU[i][j] = (v8f)0.0f; }

  const int ar = tid >> 2;            // A row 0..63
  const int ac = (tid & 3) * 16;      // A col seg: 2 x b128
  const int br = tid >> 1;            // B row 0..127
  const int bc = (tid & 1) * 32;      // B col seg: 4 x b128

  const unsigned short* aX = xb + (size_t)rows[ar] * H_DIM;
  const unsigned short* gG = wgT + ((size_t)e * I_DIM + (size_t)nt * BN + br) * H_DIM + bc;
  const unsigned short* gU = wuT + ((size_t)e * I_DIM + (size_t)nt * BN + br) * H_DIM + bc;

  // 10 async ops per thread per stage (uniform across all lanes/waves)
  auto issue = [&](int k0, int buf) {
    async_b128(lds32(&lA[buf][ar][ac]),     aX + k0 + ac);
    async_b128(lds32(&lA[buf][ar][ac + 8]), aX + k0 + ac + 8);
#pragma unroll
    for (int j = 0; j < 4; j++) {
      async_b128(lds32(&lBg[buf][br][bc + j * 8]), gG + k0 + j * 8);
      async_b128(lds32(&lBu[buf][br][bc + j * 8]), gU + k0 + j * 8);
    }
  };

  const int NIT = H_DIM / BK;   // 16
  issue(0, 0);
  for (int it = 0; it < NIT; ++it) {
    const int cur = it & 1;
    if (it + 1 < NIT) { issue((it + 1) * BK, cur ^ 1); ASYNC_WAIT(10); }
    else              { ASYNC_WAIT(0); }
    __syncthreads();

#pragma unroll
    for (int kk = 0; kk < BK; kk += 32) {
#pragma unroll
      for (int ms = 0; ms < 2; ms++) {
        Frag16 a;
        const unsigned short* ap = &lA[cur][wr * 32 + ms * 16 + l16][kk];
        a.q[0] = *reinterpret_cast<const u32x4*>(ap + lhi * 16);
        a.q[1] = *reinterpret_cast<const u32x4*>(ap + lhi * 16 + 8);
#pragma unroll
        for (int ns = 0; ns < 2; ns++) {
          const unsigned short* bgp = &lBg[cur][wc * 32 + ns * 16 + l16][kk + lhi * 16];
          const unsigned short* bup = &lBu[cur][wc * 32 + ns * 16 + l16][kk + lhi * 16];
          Frag16 bg, bu;
          bg.q[0] = *reinterpret_cast<const u32x4*>(bgp);
          bg.q[1] = *reinterpret_cast<const u32x4*>(bgp + 8);
          bu.q[0] = *reinterpret_cast<const u32x4*>(bup);
          bu.q[1] = *reinterpret_cast<const u32x4*>(bup + 8);
          accG[ms][ns] = __builtin_amdgcn_wmma_f32_16x16x32_bf16(
              false, a.v, false, bg.v, (short)0, accG[ms][ns], false, false);
          accU[ms][ns] = __builtin_amdgcn_wmma_f32_16x16x32_bf16(
              false, a.v, false, bu.v, (short)0, accU[ms][ns], false, false);
        }
      }
    }
    __syncthreads();   // protect buf[cur] before stage it+2 overwrites it
  }

  const size_t hrow0 = (size_t)offsets[e];
#pragma unroll
  for (int ms = 0; ms < 2; ms++) {
#pragma unroll
    for (int ns = 0; ns < 2; ns++) {
#pragma unroll
      for (int r = 0; r < 8; r++) {
        int ml   = lhi * 8 + r;
        int mrow = mt * BM + wr * 32 + ms * 16 + ml;
        if (mrow < Ne) {
          int ncol = nt * BN + wc * 32 + ns * 16 + l16;
          float g = accG[ms][ns][r], u = accU[ms][ns][r];
          float sv = g * fast_rcp(1.0f + __expf(-g));     // silu
          hbuf[(hrow0 + (size_t)mrow) * I_DIM + ncol] = f2bf(sv * u);
        }
      }
    }
  }
}

// ---------------------------------------------------------------- GEMM2: out += (h Wd) * we
// wdT: [E][H][I] bf16 (N-major, K contiguous). Double-buffered async pipeline.
__global__ void __launch_bounds__(256)
k_gemm2(const unsigned short* __restrict__ hbuf,
        const unsigned short* __restrict__ wdT,
        const float* __restrict__ wew,
        const int* __restrict__ counts,
        const int* __restrict__ offsets,
        const int* __restrict__ list,
        float* __restrict__ out) {
  const int e  = blockIdx.z;
  const int mt = blockIdx.x;
  const int nt = blockIdx.y;
  const int Ne = counts[e];
  if (mt * BM >= Ne) return;

  __shared__ __align__(16) unsigned short lA[2][BM][BK];
  __shared__ __align__(16) unsigned short lB[2][BN][BK];
  __shared__ int   rows[BM];
  __shared__ float rowW[BM];

  const int tid  = threadIdx.x;
  const int lane = tid & 31;
  const int w    = tid >> 5;
  const int wr   = w >> 2;
  const int wc   = w & 3;
  const int l16  = lane & 15;
  const int lhi  = lane >> 4;

  if (tid < BM) {
    int r   = mt * BM + tid;
    int tok = list[e * T_TOK + min(r, Ne - 1)];
    rows[tid] = tok;
    rowW[tid] = wew[(size_t)tok * E_NUM + e];
  }
  __syncthreads();

  v8f acc[2][2];
#pragma unroll
  for (int i = 0; i < 2; i++)
#pragma unroll
    for (int j = 0; j < 2; j++) acc[i][j] = (v8f)0.0f;

  const int ar = tid >> 2;
  const int ac = (tid & 3) * 16;
  const int br = tid >> 1;
  const int bc = (tid & 1) * 32;
  const size_t hrow0 = (size_t)offsets[e];

  const unsigned short* aH = hbuf + (hrow0 + (size_t)min(mt * BM + ar, Ne - 1)) * I_DIM;
  const unsigned short* gD = wdT + ((size_t)e * H_DIM + (size_t)nt * BN + br) * I_DIM + bc;

  // 6 async ops per thread per stage (uniform)
  auto issue = [&](int k0, int buf) {
    async_b128(lds32(&lA[buf][ar][ac]),     aH + k0 + ac);
    async_b128(lds32(&lA[buf][ar][ac + 8]), aH + k0 + ac + 8);
#pragma unroll
    for (int j = 0; j < 4; j++)
      async_b128(lds32(&lB[buf][br][bc + j * 8]), gD + k0 + j * 8);
  };

  const int NIT = I_DIM / BK;   // 44
  issue(0, 0);
  for (int it = 0; it < NIT; ++it) {
    const int cur = it & 1;
    if (it + 1 < NIT) { issue((it + 1) * BK, cur ^ 1); ASYNC_WAIT(6); }
    else              { ASYNC_WAIT(0); }
    __syncthreads();

#pragma unroll
    for (int kk = 0; kk < BK; kk += 32) {
#pragma unroll
      for (int ms = 0; ms < 2; ms++) {
        Frag16 a;
        const unsigned short* ap = &lA[cur][wr * 32 + ms * 16 + l16][kk];
        a.q[0] = *reinterpret_cast<const u32x4*>(ap + lhi * 16);
        a.q[1] = *reinterpret_cast<const u32x4*>(ap + lhi * 16 + 8);
#pragma unroll
        for (int ns = 0; ns < 2; ns++) {
          const unsigned short* bp = &lB[cur][wc * 32 + ns * 16 + l16][kk + lhi * 16];
          Frag16 b;
          b.q[0] = *reinterpret_cast<const u32x4*>(bp);
          b.q[1] = *reinterpret_cast<const u32x4*>(bp + 8);
          acc[ms][ns] = __builtin_amdgcn_wmma_f32_16x16x32_bf16(
              false, a.v, false, b.v, (short)0, acc[ms][ns], false, false);
        }
      }
    }
    __syncthreads();
  }

#pragma unroll
  for (int ms = 0; ms < 2; ms++) {
#pragma unroll
    for (int ns = 0; ns < 2; ns++) {
#pragma unroll
      for (int r = 0; r < 8; r++) {
        int ml   = lhi * 8 + r;
        int mloc = wr * 32 + ms * 16 + ml;
        int mrow = mt * BM + mloc;
        if (mrow < Ne) {
          int tok  = rows[mloc];
          int ncol = nt * BN + wc * 32 + ns * 16 + l16;
          float v  = acc[ms][ns][r] * rowW[mloc];
          unsafeAtomicAdd(&out[(size_t)tok * H_DIM + ncol], v);
        }
      }
    }
  }
}

// ---------------------------------------------------------------- host
extern "C" void kernel_launch(void* const* d_in, const int* in_sizes, int n_in,
                              void* d_out, int out_size, void* d_ws, size_t ws_size,
                              hipStream_t stream) {
  const float* x  = (const float*)d_in[0];
  const float* gw = (const float*)d_in[1];
  const float* wg = (const float*)d_in[2];
  const float* wu = (const float*)d_in[3];
  const float* wd = (const float*)d_in[4];
  float* out    = (float*)d_out;
  float* logits = out + (size_t)T_TOK * H_DIM;

  char* ws = (char*)d_ws;
  int* counts  = (int*)ws;
  int* offsets = counts + 8;
  size_t off = 256;
  unsigned short* xb  = (unsigned short*)(ws + off); off += (size_t)T_TOK * H_DIM * 2;
  unsigned short* wgT = (unsigned short*)(ws + off); off += (size_t)E_NUM * H_DIM * I_DIM * 2;
  unsigned short* wuT = (unsigned short*)(ws + off); off += (size_t)E_NUM * H_DIM * I_DIM * 2;
  unsigned short* wdT = (unsigned short*)(ws + off); off += (size_t)E_NUM * I_DIM * H_DIM * 2;
  unsigned short* hb  = (unsigned short*)(ws + off); off += (size_t)2 * T_TOK * I_DIM * 2;
  float* wew = (float*)(ws + off); off += (size_t)T_TOK * E_NUM * 4;
  int*   list = (int*)(ws + off);  off += (size_t)E_NUM * T_TOK * 4;
  (void)off; (void)ws_size; (void)in_sizes; (void)n_in; (void)out_size;

  k_init<<<2048, 256, 0, stream>>>(out, T_TOK * H_DIM, counts);
  k_cvt<<<2048, 256, 0, stream>>>(x, xb, T_TOK * H_DIM);
  {
    dim3 tg(I_DIM / 32, H_DIM / 32, E_NUM);   // [E][H][I] -> [E][I][H]
    k_cvt_tr<<<tg, 256, 0, stream>>>(wg, wgT, H_DIM, I_DIM);
    k_cvt_tr<<<tg, 256, 0, stream>>>(wu, wuT, H_DIM, I_DIM);
    dim3 td(H_DIM / 32, I_DIM / 32, E_NUM);   // [E][I][H] -> [E][H][I]
    k_cvt_tr<<<td, 256, 0, stream>>>(wd, wdT, I_DIM, H_DIM);
  }
  k_router<<<T_TOK / 8, 256, 0, stream>>>(x, gw, logits, wew, counts, list);
  k_offsets<<<1, 32, 0, stream>>>(counts, offsets);

  dim3 g1(T_TOK / BM, I_DIM / BN, E_NUM);   // 128 x 22 x 8
  k_gemm1<<<g1, 256, 0, stream>>>(xb, wgT, wuT, hb, counts, offsets, list);
  dim3 g2(T_TOK / BM, H_DIM / BN, E_NUM);   // 128 x 8 x 8
  k_gemm2<<<g2, 256, 0, stream>>>(hb, wdT, wew, counts, offsets, list, out);
}
